// MoE_48275432407261
// MI455X (gfx1250) — matrix operand order repaired and checked by reading, not compiled, not run on hardware
//
#include <hip/hip_runtime.h>
#include <hip/hip_bf16.h>

// ---- problem constants (from reference) ----
#define T_TOKENS 2048   // B*S
#define D_MODEL  2048
#define F_FF     8192
#define NE       4
#define TOPK     2
#define TM       128                             // M tile (slots)
#define SLOT_CAP (T_TOKENS * TOPK + NE * TM)     // 4608: segments rounded up to TM
#define LDA      40                              // padded LDS row (bf16) for 32-wide K tiles

typedef __bf16 bf16;
typedef __attribute__((ext_vector_type(2)))  bf16  bf16x2;
typedef __attribute__((ext_vector_type(8)))  bf16  bf16x8;
typedef __attribute__((ext_vector_type(16))) bf16  v16bf;
typedef __attribute__((ext_vector_type(8)))  float v8f;
typedef __attribute__((ext_vector_type(4)))  float f32x4;

#if __has_builtin(__builtin_amdgcn_global_load_async_to_lds_b128) && \
    __has_builtin(__builtin_amdgcn_s_wait_asynccnt)
#define HAVE_ASYNC_LDS 1
#endif

// fp32 -> bf16: native conversion (gfx1250 has hw bf16 support; backend can
// fuse pairs into v_cvt_pk_bf16_f32)
__device__ __forceinline__ bf16 f2bf(float f) {
    return (bf16)f;
}

__device__ __forceinline__ bf16x2 f2bf2(float a, float b) {
#if __has_builtin(__builtin_amdgcn_cvt_pk_bf16_f32)
    return __builtin_amdgcn_cvt_pk_bf16_f32(a, b);
#else
    bf16x2 r; r.x = (bf16)a; r.y = (bf16)b; return r;
#endif
}

__device__ __forceinline__ float fast_rcp(float x) {
#if __has_builtin(__builtin_amdgcn_rcpf)
    return __builtin_amdgcn_rcpf(x);
#else
    return 1.0f / x;
#endif
}

__device__ __forceinline__ float silu_mul(float g, float u) {
    return g * fast_rcp(1.0f + __expf(-g)) * u;
}

#ifdef HAVE_ASYNC_LDS
typedef int v4i_vs __attribute__((vector_size(16)));
typedef __attribute__((address_space(1))) v4i_vs* as1_v4i_p;
typedef __attribute__((address_space(3))) v4i_vs* as3_v4i_p;

__device__ __forceinline__ void async_cp16(const bf16* g, bf16* l) {
    // generic -> integer -> target-AS pointer (flat->local truncates per ISA)
    as1_v4i_p gp = (as1_v4i_p)(unsigned long long)g;
    as3_v4i_p lp = (as3_v4i_p)(unsigned)(unsigned long long)l;
    __builtin_amdgcn_global_load_async_to_lds_b128(gp, lp, 0, 0);
}
__device__ __forceinline__ void async_wait0() { __builtin_amdgcn_s_wait_asynccnt(0); }
#endif

__device__ __forceinline__ v16bf make_frag(bf16x8 lo, bf16x8 hi) {
    v16bf f;
#pragma unroll
    for (int i = 0; i < 8; ++i) { f[i] = lo[i]; f[i + 8] = hi[i]; }
    return f;
}

// A fragment: 16x32 bf16 from LDS tile [rows][LDA].
// lanes 0-15 (M=lane): K 0..7 then 16..23 ; lanes 16-31: K 8..15 then 24..31
__device__ __forceinline__ v16bf load_a_frag(const bf16* As, int row_base, int lane) {
    const int half = lane >> 4;
    const int r    = lane & 15;
    const bf16* p  = As + (row_base + r) * LDA + half * 8;
    return make_frag(*(const bf16x8*)p, *(const bf16x8*)(p + 16));
}

// B fragment: 32x16 bf16 from LDS tile stored transposed [n][LDA] (K contiguous).
// lanes 0-15: N=lane, K 0..7 then 16..23 ; lanes 16-31: N=lane-16, K 8..15 then 24..31
__device__ __forceinline__ v16bf load_b_frag(const bf16* Bt, int n_base, int lane) {
    const int half = lane >> 4;
    const int n    = lane & 15;
    const bf16* p  = Bt + (n_base + n) * LDA + half * 8;
    return make_frag(*(const bf16x8*)p, *(const bf16x8*)(p + 16));
}

__device__ __forceinline__ v8f wmma_bf16(v16bf a, v16bf b, v8f c) {
    return __builtin_amdgcn_wmma_f32_16x16x32_bf16(false, a, false, b, (short)0, c, false, false);
}

// ---------------- kernel 0: init ----------------
__global__ __launch_bounds__(256) void moe_init(const float* __restrict__ x,
                                                float* __restrict__ out,
                                                bf16* __restrict__ Xbf,
                                                int* __restrict__ counts,
                                                int* __restrict__ token_ids,
                                                float* __restrict__ gates) {
    const int i = blockIdx.x * 256 + threadIdx.x;
    if (i < T_TOKENS * D_MODEL) {
        out[i] = 0.0f;
        Xbf[i] = f2bf(x[i]);
    }
    if (i < SLOT_CAP) { token_ids[i] = 0; gates[i] = 0.0f; }
    if (i < NE) counts[i] = 0;
}

// ---------------- kernel 1: router (softmax + top-2) ----------------
__global__ __launch_bounds__(256) void moe_router(const float* __restrict__ x,
                                                  const float* __restrict__ Wr,
                                                  int* __restrict__ counts,
                                                  int* __restrict__ tk_idx,
                                                  float* __restrict__ tk_val) {
    const int lane = threadIdx.x & 31;
    const int wave = threadIdx.x >> 5;
    const int t = blockIdx.x * 8 + wave;
    if (t >= T_TOKENS) return;

    const float* xr = x + (size_t)t * D_MODEL;
    float acc[NE] = {0.f, 0.f, 0.f, 0.f};
    for (int d = lane; d < D_MODEL; d += 32) {
        const float xv = xr[d];
        const float* w = Wr + (size_t)d * NE;
#pragma unroll
        for (int e = 0; e < NE; ++e) acc[e] += xv * w[e];
    }
#pragma unroll
    for (int off = 16; off > 0; off >>= 1)
#pragma unroll
        for (int e = 0; e < NE; ++e) acc[e] += __shfl_xor(acc[e], off, 32);

    if (lane == 0) {
        float m = fmaxf(fmaxf(acc[0], acc[1]), fmaxf(acc[2], acc[3]));
        float p[NE], s = 0.f;
#pragma unroll
        for (int e = 0; e < NE; ++e) { p[e] = __expf(acc[e] - m); s += p[e]; }
        const float inv = 1.f / s;
#pragma unroll
        for (int e = 0; e < NE; ++e) p[e] *= inv;
        int i1 = 0;
#pragma unroll
        for (int e = 1; e < NE; ++e) if (p[e] > p[i1]) i1 = e;
        int i2 = (i1 == 0) ? 1 : 0;
#pragma unroll
        for (int e = 0; e < NE; ++e) if (e != i1 && p[e] > p[i2]) i2 = e;
        atomicAdd(&counts[i1], 1);
        atomicAdd(&counts[i2], 1);
        tk_idx[t * 2 + 0] = i1; tk_val[t * 2 + 0] = p[i1];
        tk_idx[t * 2 + 1] = i2; tk_val[t * 2 + 1] = p[i2];
    }
}

// ---------------- kernel 2: segment offsets (rounded to TM) ----------------
__global__ void moe_offsets(const int* __restrict__ counts,
                            int* __restrict__ offs, int* __restrict__ cursor) {
    if (threadIdx.x == 0 && blockIdx.x == 0) {
        int o = 0;
        for (int e = 0; e < NE; ++e) {
            offs[e] = o; cursor[e] = o;
            o += (counts[e] + TM - 1) & ~(TM - 1);
        }
        offs[NE] = o;
    }
}

// ---------------- kernel 3: scatter tokens into slot lists ----------------
__global__ __launch_bounds__(256) void moe_assign(const int* __restrict__ tk_idx,
                                                  const float* __restrict__ tk_val,
                                                  int* __restrict__ cursor,
                                                  int* __restrict__ token_ids,
                                                  float* __restrict__ gates) {
    const int t = blockIdx.x * 256 + threadIdx.x;
    if (t >= T_TOKENS) return;
#pragma unroll
    for (int k = 0; k < TOPK; ++k) {
        const int e = tk_idx[t * 2 + k];
        const int slot = atomicAdd(&cursor[e], 1);
        token_ids[slot] = t;
        gates[slot] = tk_val[t * 2 + k];
    }
}

// ---------------- kernel 4: grouped GEMM 1  H = silu(X*Wg) * (X*Wu) ----------------
// block tile 128(M) x 64(F); 8 waves = 8 M sub-tiles, each covering all 64 N
// double-buffered LDS, A tile via async global->LDS copy
__global__ __launch_bounds__(256) void moe_gemm1(const float* __restrict__ Wg,
                                                 const float* __restrict__ Wu,
                                                 const bf16* __restrict__ Xbf,
                                                 const int* __restrict__ token_ids,
                                                 const int* __restrict__ counts,
                                                 const int* __restrict__ offs,
                                                 bf16* __restrict__ H) {
    const int e = blockIdx.z;
    const int mtile = blockIdx.y;
    const int cnt = counts[e];
    if (mtile * TM >= cnt) return;
    const int slot0 = offs[e] + mtile * TM;
    const int f0 = blockIdx.x * 64;
    const float* WgE = Wg + (size_t)e * D_MODEL * F_FF;
    const float* WuE = Wu + (size_t)e * D_MODEL * F_FF;

    __shared__ bf16 As[2][TM * LDA];
    __shared__ bf16 Bgs[2][64 * LDA];
    __shared__ bf16 Bus[2][64 * LDA];
    __shared__ int  tokS[TM];

    const int tid = threadIdx.x;
    if (tid < TM) tokS[tid] = token_ids[slot0 + tid];
    __syncthreads();

    const int lane = tid & 31;
    const int wm   = tid >> 5;                 // wave id = M sub-tile (16 rows)

    v8f accg[4], accu[4];
#pragma unroll
    for (int s = 0; s < 4; ++s) { accg[s] = (v8f){0,0,0,0,0,0,0,0}; accu[s] = (v8f){0,0,0,0,0,0,0,0}; }

    // A staging: 128 rows x 32 cols bf16; thread -> row tid>>1, 16 cols
    const int arow = tid >> 1;
    const int acol = (tid & 1) * 16;
    const bf16* aG = Xbf + (size_t)tokS[arow] * D_MODEL + acol;
    // B staging: 32(K) x 64(N) fp32 per matrix; thread -> row tid>>3, 8 cols
    const int brow = tid >> 3;
    const int bcol = (tid & 7) * 8;
    const float* gG = WgE + (size_t)brow * F_FF + f0 + bcol;
    const float* uG = WuE + (size_t)brow * F_FF + f0 + bcol;

    // ---- prologue: stage k0 = 0 into buffer 0 ----
    {
        bf16* aL = &As[0][arow * LDA + acol];
#ifdef HAVE_ASYNC_LDS
        async_cp16(aG, aL);
        async_cp16(aG + 8, aL + 8);
#else
        *(bf16x8*)aL       = *(const bf16x8*)aG;
        *(bf16x8*)(aL + 8) = *(const bf16x8*)(aG + 8);
#endif
        f32x4 g0 = *(const f32x4*)gG, g1 = *(const f32x4*)(gG + 4);
        f32x4 u0 = *(const f32x4*)uG, u1 = *(const f32x4*)(uG + 4);
#pragma unroll
        for (int j = 0; j < 4; j += 2) {
            bf16x2 pg0 = f2bf2(g0[j], g0[j + 1]);
            bf16x2 pg1 = f2bf2(g1[j], g1[j + 1]);
            bf16x2 pu0 = f2bf2(u0[j], u0[j + 1]);
            bf16x2 pu1 = f2bf2(u1[j], u1[j + 1]);
            Bgs[0][(bcol + j) * LDA + brow]     = pg0.x;
            Bgs[0][(bcol + j + 1) * LDA + brow] = pg0.y;
            Bgs[0][(bcol + 4 + j) * LDA + brow]     = pg1.x;
            Bgs[0][(bcol + 4 + j + 1) * LDA + brow] = pg1.y;
            Bus[0][(bcol + j) * LDA + brow]     = pu0.x;
            Bus[0][(bcol + j + 1) * LDA + brow] = pu0.y;
            Bus[0][(bcol + 4 + j) * LDA + brow]     = pu1.x;
            Bus[0][(bcol + 4 + j + 1) * LDA + brow] = pu1.y;
        }
#ifdef HAVE_ASYNC_LDS
        async_wait0();
#endif
        __syncthreads();
    }

    int buf = 0;
    for (int k0 = 0; k0 < D_MODEL; k0 += 32) {
        const int  nxt  = buf ^ 1;
        const bool more = (k0 + 32) < D_MODEL;
        f32x4 g0, g1, u0, u1;
#ifndef HAVE_ASYNC_LDS
        bf16x8 a0, a1;
#endif
        if (more) {
            const bf16* aGk = aG + k0 + 32;
#ifdef HAVE_ASYNC_LDS
            bf16* aL = &As[nxt][arow * LDA + acol];
            async_cp16(aGk, aL);
            async_cp16(aGk + 8, aL + 8);
#else
            a0 = *(const bf16x8*)aGk;
            a1 = *(const bf16x8*)(aGk + 8);
#endif
            const float* pg = gG + (size_t)(k0 + 32) * F_FF;
            const float* pu = uG + (size_t)(k0 + 32) * F_FF;
            g0 = *(const f32x4*)pg; g1 = *(const f32x4*)(pg + 4);
            u0 = *(const f32x4*)pu; u1 = *(const f32x4*)(pu + 4);
            __builtin_prefetch(pg + 32 * (size_t)F_FF, 0, 3);
            __builtin_prefetch(pu + 32 * (size_t)F_FF, 0, 3);
        }

        // ---- compute on current buffer ----
        const v16bf a = load_a_frag(As[buf], wm * 16, lane);
#pragma unroll
        for (int s = 0; s < 4; ++s) {
            const v16bf bg = load_b_frag(Bgs[buf], s * 16, lane);
            accg[s] = wmma_bf16(a, bg, accg[s]);
        }
#pragma unroll
        for (int s = 0; s < 4; ++s) {
            const v16bf bu = load_b_frag(Bus[buf], s * 16, lane);
            accu[s] = wmma_bf16(a, bu, accu[s]);
        }

        if (more) {
#ifndef HAVE_ASYNC_LDS
            bf16* aL = &As[nxt][arow * LDA + acol];
            *(bf16x8*)aL       = a0;
            *(bf16x8*)(aL + 8) = a1;
#endif
#pragma unroll
            for (int j = 0; j < 4; j += 2) {
                bf16x2 pg0 = f2bf2(g0[j], g0[j + 1]);
                bf16x2 pg1 = f2bf2(g1[j], g1[j + 1]);
                bf16x2 pu0 = f2bf2(u0[j], u0[j + 1]);
                bf16x2 pu1 = f2bf2(u1[j], u1[j + 1]);
                Bgs[nxt][(bcol + j) * LDA + brow]     = pg0.x;
                Bgs[nxt][(bcol + j + 1) * LDA + brow] = pg0.y;
                Bgs[nxt][(bcol + 4 + j) * LDA + brow]     = pg1.x;
                Bgs[nxt][(bcol + 4 + j + 1) * LDA + brow] = pg1.y;
                Bus[nxt][(bcol + j) * LDA + brow]     = pu0.x;
                Bus[nxt][(bcol + j + 1) * LDA + brow] = pu0.y;
                Bus[nxt][(bcol + 4 + j) * LDA + brow]     = pu1.x;
                Bus[nxt][(bcol + 4 + j + 1) * LDA + brow] = pu1.y;
            }
#ifdef HAVE_ASYNC_LDS
            async_wait0();
#endif
        }
        __syncthreads();
        buf = nxt;
    }

    // ---- epilogue: h = silu(g)*u -> bf16 H[slot][f] ----
    const int n   = lane & 15;
    const int mhi = (lane >> 4) * 8;
#pragma unroll
    for (int s = 0; s < 4; ++s) {
        const int fcol = f0 + s * 16 + n;
        bf16* hp = H + (size_t)(slot0 + wm * 16 + mhi) * F_FF + fcol;
#pragma unroll
        for (int i = 0; i < 8; i += 2) {
            const float h0 = silu_mul(accg[s][i], accu[s][i]);
            const float h1 = silu_mul(accg[s][i + 1], accu[s][i + 1]);
            const bf16x2 pk = f2bf2(h0, h1);
            hp[(size_t)i * F_FF]       = pk.x;
            hp[(size_t)(i + 1) * F_FF] = pk.y;
        }
    }
}

// ---------------- kernel 5: grouped GEMM 2  out += gate * (H * Wd) ----------------
__global__ __launch_bounds__(256) void moe_gemm2(const float* __restrict__ Wd,
                                                 const bf16* __restrict__ H,
                                                 const int* __restrict__ token_ids,
                                                 const float* __restrict__ gates,
                                                 const int* __restrict__ counts,
                                                 const int* __restrict__ offs,
                                                 float* __restrict__ out) {
    const int e = blockIdx.z;
    const int mtile = blockIdx.y;
    const int cnt = counts[e];
    if (mtile * TM >= cnt) return;
    const int slot0 = offs[e] + mtile * TM;
    const int d0 = blockIdx.x * 64;
    const float* WdE = Wd + (size_t)e * F_FF * D_MODEL;

    __shared__ bf16  As[2][TM * LDA];
    __shared__ bf16  Bts[2][64 * LDA];
    __shared__ int   tokS[TM];
    __shared__ float gateS[TM];

    const int tid = threadIdx.x;
    if (tid < TM) { tokS[tid] = token_ids[slot0 + tid]; gateS[tid] = gates[slot0 + tid]; }
    __syncthreads();

    const int lane = tid & 31;
    const int wm   = tid >> 5;

    v8f acc[4];
#pragma unroll
    for (int s = 0; s < 4; ++s) acc[s] = (v8f){0,0,0,0,0,0,0,0};

    const int arow = tid >> 1;
    const int acol = (tid & 1) * 16;
    const bf16* aG = H + (size_t)(slot0 + arow) * F_FF + acol;
    const int brow = tid >> 3;
    const int bcol = (tid & 7) * 8;
    const float* bG = WdE + (size_t)brow * D_MODEL + d0 + bcol;

    // ---- prologue ----
    {
        bf16* aL = &As[0][arow * LDA + acol];
#ifdef HAVE_ASYNC_LDS
        async_cp16(aG, aL);
        async_cp16(aG + 8, aL + 8);
#else
        *(bf16x8*)aL       = *(const bf16x8*)aG;
        *(bf16x8*)(aL + 8) = *(const bf16x8*)(aG + 8);
#endif
        f32x4 b0 = *(const f32x4*)bG, b1 = *(const f32x4*)(bG + 4);
#pragma unroll
        for (int j = 0; j < 4; j += 2) {
            bf16x2 p0 = f2bf2(b0[j], b0[j + 1]);
            bf16x2 p1 = f2bf2(b1[j], b1[j + 1]);
            Bts[0][(bcol + j) * LDA + brow]     = p0.x;
            Bts[0][(bcol + j + 1) * LDA + brow] = p0.y;
            Bts[0][(bcol + 4 + j) * LDA + brow]     = p1.x;
            Bts[0][(bcol + 4 + j + 1) * LDA + brow] = p1.y;
        }
#ifdef HAVE_ASYNC_LDS
        async_wait0();
#endif
        __syncthreads();
    }

    int buf = 0;
    for (int k0 = 0; k0 < F_FF; k0 += 32) {
        const int  nxt  = buf ^ 1;
        const bool more = (k0 + 32) < F_FF;
        f32x4 b0, b1;
#ifndef HAVE_ASYNC_LDS
        bf16x8 a0, a1;
#endif
        if (more) {
            const bf16* aGk = aG + k0 + 32;
#ifdef HAVE_ASYNC_LDS
            bf16* aL = &As[nxt][arow * LDA + acol];
            async_cp16(aGk, aL);
            async_cp16(aGk + 8, aL + 8);
#else
            a0 = *(const bf16x8*)aGk;
            a1 = *(const bf16x8*)(aGk + 8);
#endif
            const float* pb = bG + (size_t)(k0 + 32) * D_MODEL;
            b0 = *(const f32x4*)pb; b1 = *(const f32x4*)(pb + 4);
            __builtin_prefetch(pb + 32 * (size_t)D_MODEL, 0, 3);
        }

        const v16bf a = load_a_frag(As[buf], wm * 16, lane);
#pragma unroll
        for (int s = 0; s < 4; ++s) {
            const v16bf b = load_b_frag(Bts[buf], s * 16, lane);
            acc[s] = wmma_bf16(a, b, acc[s]);
        }

        if (more) {
#ifndef HAVE_ASYNC_LDS
            bf16* aL = &As[nxt][arow * LDA + acol];
            *(bf16x8*)aL       = a0;
            *(bf16x8*)(aL + 8) = a1;
#endif
#pragma unroll
            for (int j = 0; j < 4; j += 2) {
                bf16x2 p0 = f2bf2(b0[j], b0[j + 1]);
                bf16x2 p1 = f2bf2(b1[j], b1[j + 1]);
                Bts[nxt][(bcol + j) * LDA + brow]     = p0.x;
                Bts[nxt][(bcol + j + 1) * LDA + brow] = p0.y;
                Bts[nxt][(bcol + 4 + j) * LDA + brow]     = p1.x;
                Bts[nxt][(bcol + 4 + j + 1) * LDA + brow] = p1.y;
            }
#ifdef HAVE_ASYNC_LDS
            async_wait0();
#endif
        }
        __syncthreads();
        buf = nxt;
    }

    // ---- epilogue: out[token] += gate * acc ----
    const int n   = lane & 15;
    const int mhi = (lane >> 4) * 8;
#pragma unroll
    for (int s = 0; s < 4; ++s) {
        const int dcol = d0 + s * 16 + n;
#pragma unroll
        for (int i = 0; i < 8; ++i) {
            const int mloc = wm * 16 + mhi + i;
            const float v = gateS[mloc] * acc[s][i];
            atomicAdd(out + (size_t)tokS[mloc] * D_MODEL + dcol, v);
        }
    }
}

// ---------------- launch ----------------
extern "C" void kernel_launch(void* const* d_in, const int* in_sizes, int n_in,
                              void* d_out, int out_size, void* d_ws, size_t ws_size,
                              hipStream_t stream) {
    (void)in_sizes; (void)n_in; (void)out_size; (void)ws_size;
    const float* x  = (const float*)d_in[0];
    const float* Wr = (const float*)d_in[1];
    const float* Wg = (const float*)d_in[2];
    const float* Wu = (const float*)d_in[3];
    const float* Wd = (const float*)d_in[4];
    float* out = (float*)d_out;

    char* w = (char*)d_ws;
    int*   counts    = (int*)(w + 0);
    int*   offs      = (int*)(w + 256);
    int*   cursor    = (int*)(w + 512);
    int*   tk_idx    = (int*)(w + 4096);
    float* tk_val    = (float*)(w + 4096 + 16384);
    int*   token_ids = (int*)(w + 36864);
    float* gates     = (float*)(w + 69632);
    bf16*  Xbf       = (bf16*)(w + 131072);
    bf16*  H         = (bf16*)(w + 131072 + (size_t)T_TOKENS * D_MODEL * 2);

    const int nInit = (T_TOKENS * D_MODEL + 255) / 256;
    moe_init<<<nInit, 256, 0, stream>>>(x, out, Xbf, counts, token_ids, gates);
    moe_router<<<T_TOKENS / 8, 256, 0, stream>>>(x, Wr, counts, tk_idx, tk_val);
    moe_offsets<<<1, 32, 0, stream>>>(counts, offs, cursor);
    moe_assign<<<(T_TOKENS + 255) / 256, 256, 0, stream>>>(tk_idx, tk_val, cursor, token_ids, gates);
    moe_gemm1<<<dim3(F_FF / 64, T_TOKENS / TM, NE), 256, 0, stream>>>(
        Wg, Wu, Xbf, token_ids, counts, offs, H);
    moe_gemm2<<<dim3(D_MODEL / 64, T_TOKENS / TM, NE), 256, 0, stream>>>(
        Wd, H, token_ids, gates, counts, offs, out);
}
